// GptOssDFlashDecoderLayer_22952305230272
// MI455X (gfx1250) — matrix operand order repaired and checked by reading, not compiled, not run on hardware
//
#include <hip/hip_runtime.h>
#include <hip/hip_bf16.h>
#include <math.h>
#include <stdint.h>

// ---------------------------------------------------------------- constants
constexpr int   Qn   = 2048;
constexpr int   CTXn = 2048;
constexpr int   Sn   = CTXn + Qn;   // 4096 keys
constexpr int   Hn   = 2048;
constexpr int   NHn  = 32;
constexpr int   NKVn = 8;
constexpr int   HDn  = 64;
constexpr int   In   = 8192;
constexpr int   Dq   = NHn * HDn;   // 2048
constexpr int   Dkv  = NKVn * HDn;  // 512
constexpr float SCALEf = 0.125f;    // 64^-0.5
constexpr float EPSf   = 1e-6f;

// ---------------------------------------------------------------- WMMA types
typedef _Float16 v16h __attribute__((ext_vector_type(16)));
typedef _Float16 v8h  __attribute__((ext_vector_type(8)));
typedef float    v8f  __attribute__((ext_vector_type(8)));

union AFrag { v16h v; v8h h[2]; };

__device__ __forceinline__ v8f wmma_f16(const v16h& a, const v16h& b, const v8f& c) {
    return __builtin_amdgcn_wmma_f32_16x16x32_f16(false, a, false, b, (short)0, c,
                                                  false, false);
}

__device__ __forceinline__ float half16_max(float v) {
    #pragma unroll
    for (int m = 1; m < 16; m <<= 1) v = fmaxf(v, __shfl_xor(v, m, 32));
    return v;
}
__device__ __forceinline__ float half16_sum(float v) {
    #pragma unroll
    for (int m = 1; m < 16; m <<= 1) v += __shfl_xor(v, m, 32);
    return v;
}

// CDNA5 async copy: global -> LDS, 16 bytes per lane, tracked by ASYNCcnt.
__device__ __forceinline__ void async_copy16(uint32_t lds_off, const void* gptr) {
    unsigned long long ga = (unsigned long long)gptr;
    asm volatile("global_load_async_to_lds_b128 %0, %1, off"
                 :: "v"(lds_off), "v"(ga) : "memory");
}
__device__ __forceinline__ void wait_async0() {
    asm volatile("s_wait_asynccnt 0x0" ::: "memory");
}

// ---------------------------------------------------------------- casts
__global__ void cast_f32_f16(const float* __restrict__ in, _Float16* __restrict__ out,
                             long n) {
    long i = (long)blockIdx.x * blockDim.x + threadIdx.x;
    long stride = (long)gridDim.x * blockDim.x;
    for (; i < n; i += stride) out[i] = (_Float16)in[i];
}

// ---------------------------------------------------------------- RMSNorm (f32 in, f16 out)
__global__ __launch_bounds__(256)
void rmsnorm_f16(const float* __restrict__ X, const float* __restrict__ W,
                 _Float16* __restrict__ Y, int Hd) {
    __shared__ float red[256];
    const int row = blockIdx.x;
    const float* x = X + (size_t)row * Hd;
    float p = 0.f;
    for (int i = threadIdx.x; i < Hd; i += 256) { float v = x[i]; p += v * v; }
    red[threadIdx.x] = p;
    __syncthreads();
    for (int s = 128; s > 0; s >>= 1) {
        if (threadIdx.x < s) red[threadIdx.x] += red[threadIdx.x + s];
        __syncthreads();
    }
    const float rs = rsqrtf(red[0] / (float)Hd + EPSf);
    _Float16* y = Y + (size_t)row * Hd;
    for (int i = threadIdx.x; i < Hd; i += 256)
        y[i] = (_Float16)(x[i] * rs * W[i]);
}

// ---------------------------------------------------------------- generic WMMA GEMM
// C[M,N] = A[M,K] (f16, row-major) * W[N,K]^T (f16, row-major)
// flags: 1 = write f32 Cf, 2 = write f16 Cb, 4 = add f32 residual Res
__global__ __launch_bounds__(256)
void gemm_f16_wmma(const _Float16* __restrict__ A, const _Float16* __restrict__ Bw,
                   float* __restrict__ Cf, _Float16* __restrict__ Cb,
                   const float* __restrict__ Res,
                   int M, int N, int K, int flags) {
    const int lane  = threadIdx.x & 31;
    const int wave  = threadIdx.x >> 5;
    const int waveM = wave & 3;   // 4 waves along M
    const int waveN = wave >> 2;  // 2 waves along N
    const int m0    = blockIdx.y * 128 + waveM * 32;
    const int n0    = blockIdx.x * 128 + waveN * 64;
    const int half  = lane >> 4;
    const int l15   = lane & 15;

    v8f acc[2][4] = {};

    for (int k0 = 0; k0 < K; k0 += 32) {
        AFrag af[2];
        #pragma unroll
        for (int mt = 0; mt < 2; ++mt) {
            const _Float16* pa = A + (size_t)(m0 + mt * 16 + l15) * K + k0 + half * 16;
            af[mt].h[0] = *(const v8h*)(pa);
            af[mt].h[1] = *(const v8h*)(pa + 8);
            __builtin_prefetch(pa + 32);
        }
        #pragma unroll
        for (int nt = 0; nt < 4; ++nt) {
            const _Float16* pb = Bw + (size_t)(n0 + nt * 16 + l15) * K + k0 + half * 16;
            v16h bf = *(const v16h*)(pb);
            __builtin_prefetch(pb + 32);
            #pragma unroll
            for (int mt = 0; mt < 2; ++mt)
                acc[mt][nt] = wmma_f16(af[mt].v, bf, acc[mt][nt]);
        }
    }

    #pragma unroll
    for (int mt = 0; mt < 2; ++mt)
        #pragma unroll
        for (int nt = 0; nt < 4; ++nt)
            #pragma unroll
            for (int r = 0; r < 8; ++r) {
                const int row = m0 + mt * 16 + half * 8 + r;
                const int col = n0 + nt * 16 + l15;
                if (row < M && col < N) {
                    float v = acc[mt][nt][r];
                    if (flags & 4) v += Res[(size_t)row * N + col];
                    if (flags & 1) Cf[(size_t)row * N + col] = v;
                    if (flags & 2) Cb[(size_t)row * N + col] = (_Float16)v;
                }
            }
}

// ---------------------------------------------------------------- RoPE for Q (folds softmax scale)
// qf: [Q, NH*HD] f32 -> Qr: [NH][Q][HD] f16
__global__ void rope_q(const float* __restrict__ qf, const float* __restrict__ cosT,
                       const float* __restrict__ sinT, _Float16* __restrict__ Qr) {
    const int d  = threadIdx.x;    // 0..63
    const int h  = blockIdx.x;     // head
    const int qi = blockIdx.y;     // q row
    const int pos = CTXn + qi;
    const float c = cosT[(size_t)pos * HDn + d];
    const float s = sinT[(size_t)pos * HDn + d];
    const float* qr = qf + (size_t)qi * Dq + h * HDn;
    const float v  = qr[d];
    const float vr = (d < HDn / 2) ? -qr[d + HDn / 2] : qr[d - HDn / 2];
    Qr[((size_t)h * Qn + qi) * HDn + d] = (_Float16)((v * c + vr * s) * SCALEf);
}

// ---------------------------------------------------------------- RoPE for K + V transpose
// kcat/vcat: [S, NKV*HD] f32 -> Kr: [NKV][S][HD] f16, Vt: [NKV][HD][S] f16
__global__ void rope_k_v(const float* __restrict__ kcat, const float* __restrict__ vcat,
                         const float* __restrict__ cosT, const float* __restrict__ sinT,
                         _Float16* __restrict__ Kr, _Float16* __restrict__ Vt) {
    const int d  = threadIdx.x;    // 0..63
    const int h  = blockIdx.x;     // kv head
    const int si = blockIdx.y;     // key row
    const float c = cosT[(size_t)si * HDn + d];
    const float s = sinT[(size_t)si * HDn + d];
    const float* kr = kcat + (size_t)si * Dkv + h * HDn;
    const float v  = kr[d];
    const float vr = (d < HDn / 2) ? -kr[d + HDn / 2] : kr[d - HDn / 2];
    Kr[((size_t)h * Sn + si) * HDn + d] = (_Float16)(v * c + vr * s);
    Vt[((size_t)h * HDn + d) * Sn + si] =
        (_Float16)vcat[(size_t)si * Dkv + h * HDn + d];
}

// ---------------------------------------------------------------- flash attention (WMMA, online softmax)
// Qr:[NH][Q][64] (pre-scaled), Kr:[NKV][S][64], Vt:[NKV][64][S] -> Out:[Q, NH*64] f16
// All 4 waves share the same kv-head and key block, so K/V tiles are staged
// into LDS once per block via CDNA5 async global->LDS copies (ASYNCcnt).
__global__ __launch_bounds__(128)
void flash_attn(const _Float16* __restrict__ Qr, const _Float16* __restrict__ Kr,
                const _Float16* __restrict__ Vt, _Float16* __restrict__ Out) {
    __shared__ __align__(64) _Float16 Kt[32][64];     // [key][d]     4 KB
    __shared__ __align__(64) _Float16 Vs[64][32];     // [d][key]     4 KB
    __shared__ __align__(64) _Float16 Pl[4][16][32];  // per-wave P   4 KB

    const int lane = threadIdx.x & 31;
    const int wave = threadIdx.x >> 5;
    const int half = lane >> 4;
    const int l15  = lane & 15;
    const int h    = blockIdx.x;
    const int kvh  = h >> 2;                          // NH/NKV = 4
    const int q0   = blockIdx.y * 64 + wave * 16;

    // Preload Q A-fragments (two 16x32 K-slices covering HD=64)
    AFrag aq[2];
    #pragma unroll
    for (int ks = 0; ks < 2; ++ks) {
        const _Float16* p = Qr + ((size_t)h * Qn + q0 + l15) * HDn + ks * 32 + half * 16;
        aq[ks].h[0] = *(const v8h*)p;
        aq[ks].h[1] = *(const v8h*)(p + 8);
    }

    v8f   o[4] = {};
    float mrow[8], lrow[8];
    #pragma unroll
    for (int r = 0; r < 8; ++r) { mrow[r] = -1e30f; lrow[r] = 0.f; }

    const int t = threadIdx.x;  // 0..127

    for (int kb = 0; kb < Sn / 32; ++kb) {
        // ---- stage K (32x64) and V^T (64x32) tiles into LDS asynchronously
        #pragma unroll
        for (int i = 0; i < 2; ++i) {
            const int c = t + 128 * i;            // 0..255 chunk id (16B chunks)
            const int key = c >> 3, dc = c & 7;   // K tile: 8 chunks per key row
            async_copy16((uint32_t)(uintptr_t)&Kt[key][dc * 8],
                         Kr + ((size_t)kvh * Sn + kb * 32 + key) * HDn + dc * 8);
            const int dv = c >> 2, kc = c & 3;    // V tile: 4 chunks per d row
            async_copy16((uint32_t)(uintptr_t)&Vs[dv][kc * 8],
                         Vt + ((size_t)kvh * HDn + dv) * Sn + kb * 32 + kc * 8);
        }
        wait_async0();
        __syncthreads();

        // ---- S = Q * K^T for 32 keys (two 16-wide column tiles), from LDS
        v8f s0 = {}, s1 = {};
        #pragma unroll
        for (int ks = 0; ks < 2; ++ks) {
            v16h b0 = *(const v16h*)&Kt[l15][ks * 32 + half * 16];
            s0 = wmma_f16(aq[ks].v, b0, s0);
            v16h b1 = *(const v16h*)&Kt[16 + l15][ks * 32 + half * 16];
            s1 = wmma_f16(aq[ks].v, b1, s1);
        }

        // ---- online softmax: row = half*8 + r, 16 columns across half-wave
        #pragma unroll
        for (int r = 0; r < 8; ++r) {
            float lm = half16_max(fmaxf(s0[r], s1[r]));
            float mn = fmaxf(mrow[r], lm);
            float alpha = __expf(mrow[r] - mn);
            mrow[r] = mn;
            float p0 = __expf(s0[r] - mn);
            float p1 = __expf(s1[r] - mn);
            lrow[r] = lrow[r] * alpha + half16_sum(p0 + p1);
            #pragma unroll
            for (int nt = 0; nt < 4; ++nt) o[nt][r] *= alpha;
            Pl[wave][half * 8 + r][l15]      = (_Float16)p0;
            Pl[wave][half * 8 + r][16 + l15] = (_Float16)p1;
        }
        __syncthreads();

        // ---- reload P as 16x32 A-fragment
        AFrag pa;
        const _Float16* pp = &Pl[wave][l15][half * 16];
        pa.h[0] = *(const v8h*)pp;
        pa.h[1] = *(const v8h*)(pp + 8);

        // ---- O += P * V (V^T rows contiguous in keys), from LDS
        #pragma unroll
        for (int nt = 0; nt < 4; ++nt) {
            v16h bv = *(const v16h*)&Vs[nt * 16 + l15][half * 16];
            o[nt] = wmma_f16(pa.v, bv, o[nt]);
        }
        __syncthreads();  // all reads of Kt/Vs/Pl done before next stage
    }

    #pragma unroll
    for (int nt = 0; nt < 4; ++nt)
        #pragma unroll
        for (int r = 0; r < 8; ++r) {
            const int q = q0 + half * 8 + r;
            const int d = nt * 16 + l15;
            Out[(size_t)q * Dq + h * HDn + d] = (_Float16)(o[nt][r] / lrow[r]);
        }
}

// ---------------------------------------------------------------- SiLU(gate) * up
__global__ void silu_mul(const _Float16* __restrict__ gu, _Float16* __restrict__ mlp) {
    long i = (long)blockIdx.x * blockDim.x + threadIdx.x;
    const long n = (long)Qn * In;
    const long stride = (long)gridDim.x * blockDim.x;
    for (; i < n; i += stride) {
        const long q = i / In;
        const long c = i - q * In;
        float g = (float)gu[(size_t)q * 2 * In + c];
        float u = (float)gu[(size_t)q * 2 * In + In + c];
        mlp[i] = (_Float16)(g / (1.f + __expf(-g)) * u);
    }
}

// ---------------------------------------------------------------- launch
extern "C" void kernel_launch(void* const* d_in, const int* in_sizes, int n_in,
                              void* d_out, int out_size, void* d_ws, size_t ws_size,
                              hipStream_t stream) {
    const float* hidden = (const float*)d_in[0];   // [Q, H]
    const float* target = (const float*)d_in[1];   // [CTX, H]
    const float* cosT   = (const float*)d_in[2];   // [S, HD]
    const float* sinT   = (const float*)d_in[3];   // [S, HD]
    const float* ln1w   = (const float*)d_in[4];
    const float* Wq     = (const float*)d_in[5];   // [Dq, H]
    const float* Wk     = (const float*)d_in[6];   // [Dkv, H]
    const float* Wv     = (const float*)d_in[7];   // [Dkv, H]
    const float* Wo     = (const float*)d_in[8];   // [H, Dq]
    const float* ln2w   = (const float*)d_in[9];
    const float* Wgu    = (const float*)d_in[10];  // [2I, H]
    const float* Wd     = (const float*)d_in[11];  // [H, I]
    float* out = (float*)d_out;                    // [Q, H]

    char*  ws  = (char*)d_ws;
    size_t off = 0;
    auto alloc = [&](size_t bytes) -> void* {
        void* p = ws + off;
        off = (off + bytes + 255) & ~(size_t)255;
        return p;
    };

    _Float16* tb   = (_Float16*)alloc((size_t)CTXn * Hn * 2);
    _Float16* xb   = (_Float16*)alloc((size_t)Qn * Hn * 2);
    _Float16* Wqh  = (_Float16*)alloc((size_t)Dq * Hn * 2);
    _Float16* Wkh  = (_Float16*)alloc((size_t)Dkv * Hn * 2);
    _Float16* Wvh  = (_Float16*)alloc((size_t)Dkv * Hn * 2);
    _Float16* Woh  = (_Float16*)alloc((size_t)Hn * Dq * 2);
    _Float16* Wguh = (_Float16*)alloc((size_t)2 * In * Hn * 2);
    _Float16* Wdh  = (_Float16*)alloc((size_t)Hn * In * 2);
    float*    qf   = (float*)alloc((size_t)Qn * Dq * 4);
    float*    kcat = (float*)alloc((size_t)Sn * Dkv * 4);
    float*    vcat = (float*)alloc((size_t)Sn * Dkv * 4);
    _Float16* Qr   = (_Float16*)alloc((size_t)NHn * Qn * HDn * 2);
    _Float16* Kr   = (_Float16*)alloc((size_t)NKVn * Sn * HDn * 2);
    _Float16* Vt   = (_Float16*)alloc((size_t)NKVn * HDn * Sn * 2);
    _Float16* attn = (_Float16*)alloc((size_t)Qn * Dq * 2);
    float*    hbuf = (float*)alloc((size_t)Qn * Hn * 4);
    _Float16* yb   = (_Float16*)alloc((size_t)Qn * Hn * 2);
    _Float16* gub  = (_Float16*)alloc((size_t)Qn * 2 * In * 2);
    _Float16* mlpb = (_Float16*)alloc((size_t)Qn * In * 2);

    auto cast = [&](const float* src, _Float16* dst, long n) {
        int blocks = (int)((n + 1023) / 1024);
        if (blocks > 65520) blocks = 65520;
        cast_f32_f16<<<blocks, 256, 0, stream>>>(src, dst, n);
    };

    // weight + context casts (f32 -> f16)
    cast(target, tb, (long)CTXn * Hn);
    cast(Wq, Wqh, (long)Dq * Hn);
    cast(Wk, Wkh, (long)Dkv * Hn);
    cast(Wv, Wvh, (long)Dkv * Hn);
    cast(Wo, Woh, (long)Hn * Dq);
    cast(Wgu, Wguh, (long)2 * In * Hn);
    cast(Wd, Wdh, (long)Hn * In);

    // x = rmsnorm(hidden)
    rmsnorm_f16<<<Qn, 256, 0, stream>>>(hidden, ln1w, xb, Hn);

    dim3 blk(256);
    auto gemm = [&](const _Float16* A, const _Float16* B, float* Cf, _Float16* Cb,
                    const float* Res, int M, int N, int K, int flags) {
        dim3 grid(N / 128, M / 128);
        gemm_f16_wmma<<<grid, blk, 0, stream>>>(A, B, Cf, Cb, Res, M, N, K, flags);
    };

    // projections
    gemm(xb, Wqh, qf, nullptr, nullptr, Qn, Dq, Hn, 1);
    gemm(tb, Wkh, kcat, nullptr, nullptr, CTXn, Dkv, Hn, 1);
    gemm(xb, Wkh, kcat + (size_t)CTXn * Dkv, nullptr, nullptr, Qn, Dkv, Hn, 1);
    gemm(tb, Wvh, vcat, nullptr, nullptr, CTXn, Dkv, Hn, 1);
    gemm(xb, Wvh, vcat + (size_t)CTXn * Dkv, nullptr, nullptr, Qn, Dkv, Hn, 1);

    // RoPE + layout
    rope_q<<<dim3(NHn, Qn), 64, 0, stream>>>(qf, cosT, sinT, Qr);
    rope_k_v<<<dim3(NKVn, Sn), 64, 0, stream>>>(kcat, vcat, cosT, sinT, Kr, Vt);

    // attention
    flash_attn<<<dim3(NHn, Qn / 64), 128, 0, stream>>>(Qr, Kr, Vt, attn);

    // h = hidden + attn @ Wo^T
    gemm(attn, Woh, hbuf, nullptr, hidden, Qn, Hn, Dq, 1 | 4);

    // y = rmsnorm(h)
    rmsnorm_f16<<<Qn, 256, 0, stream>>>(hbuf, ln2w, yb, Hn);

    // gu = y @ Wgu^T ; mlp = silu(gate)*up
    gemm(yb, Wguh, nullptr, gub, nullptr, Qn, 2 * In, Hn, 2);
    silu_mul<<<4096, 256, 0, stream>>>(gub, mlpb);

    // out = h + mlp @ Wd^T
    gemm(mlpb, Wdh, out, nullptr, hbuf, Qn, Hn, In, 1 | 4);
}
